// AsymmetricTemporalAttention_38938173505696
// MI455X (gfx1250) — matrix operand order repaired and checked by reading, not compiled, not run on hardware
//
#include <hip/hip_runtime.h>
#include <hip/hip_bf16.h>
#include <math.h>

typedef __bf16 bf16;
typedef __attribute__((ext_vector_type(16))) __bf16 v16bf;
typedef __attribute__((ext_vector_type(8)))  __bf16 v8bf;
typedef __attribute__((ext_vector_type(4)))  __bf16 v4bf;
typedef __attribute__((ext_vector_type(8)))  float  v8f;

#define B_   2
#define S_   2048
#define D_   1024
#define H_   16
#define DH_  64
#define ROWS (B_ * S_)   // 4096

// ---------------------------------------------------------------------------
// WMMA helpers (CDNA5 gfx1250, wave32). D = A(16x32 bf16) x B(32x16 bf16) + C(16x16 f32)
// ---------------------------------------------------------------------------
__device__ __forceinline__ v8f wmma_bf(v16bf a, v16bf b, v8f c) {
  // 8 args: (neg_a, A, neg_b, B, c_mod, C, reuse_a, reuse_b)
  return __builtin_amdgcn_wmma_f32_16x16x32_bf16(false, a, false, b, (short)0, c,
                                                 false, false);
}

// A-fragment (16x32, M x K).  Layout: lanes 0-15 row M=lane hold K=0..15,
// lanes 16-31 row M=lane-16 hold K=16..31.  Element (m,k) at base + m*ld + k.
__device__ __forceinline__ v16bf load_A_frag(const bf16* base, int ld) {
  int lane = threadIdx.x & 31;
  int row  = lane & 15;
  int half = lane >> 4;
  const bf16* p = base + row * ld + half * 16;
  v8bf lo = *(const v8bf*)(p);        // K = half*16 .. +7
  v8bf hi = *(const v8bf*)(p + 8);    // K = half*16 + 8 .. +15
  v16bf a;
#pragma unroll
  for (int i = 0; i < 8; ++i) { a[i] = lo[i]; a[i + 8] = hi[i]; }
  return a;
}

// B-fragment (32x16, K x N). Lane (n + 16*kh) element e holds B[kh*16+e][n].
// Caller passes a "transposed view": element (k,n) lives at base + n*ld + k.
__device__ __forceinline__ v16bf load_B_frag(const bf16* base, int ld) {
  int lane = threadIdx.x & 31;
  int n  = lane & 15;
  int kh = lane >> 4;
  return *(const v16bf*)(base + n * ld + kh * 16);
}

// One pipeline stage of the 16x64 GEMM tile: A fragment + 4 B fragments.
struct Frags {
  v16bf a;
  v16bf b[4];
};

__device__ __forceinline__ Frags load_frags(const bf16* abase, const bf16* w,
                                            int ncol, int k0) {
  Frags f;
  f.a = load_A_frag(abase + k0, D_);
#pragma unroll
  for (int t = 0; t < 4; ++t)
    f.b[t] = load_B_frag(w + (size_t)(ncol + t * 16) * D_ + k0, D_);
  return f;
}

// ---------------------------------------------------------------------------
// Kernel 1: convert the four fp32 weight matrices to bf16 (once per call).
// ---------------------------------------------------------------------------
__global__ __launch_bounds__(256) void cvtw_kernel(
    const float* __restrict__ a, const float* __restrict__ b,
    const float* __restrict__ c, const float* __restrict__ d,
    bf16* __restrict__ oa, bf16* __restrict__ ob,
    bf16* __restrict__ oc, bf16* __restrict__ od) {
  int i = blockIdx.x * blockDim.x + threadIdx.x;   // 0 .. D*D/4-1
  const float4* As = (const float4*)a; const float4* Bs = (const float4*)b;
  const float4* Cs = (const float4*)c; const float4* Ds = (const float4*)d;
  float4 va = As[i], vb = Bs[i], vc = Cs[i], vd = Ds[i];
  v4bf wa, wb, wc, wd;
  wa[0]=(bf16)va.x; wa[1]=(bf16)va.y; wa[2]=(bf16)va.z; wa[3]=(bf16)va.w;
  wb[0]=(bf16)vb.x; wb[1]=(bf16)vb.y; wb[2]=(bf16)vb.z; wb[3]=(bf16)vb.w;
  wc[0]=(bf16)vc.x; wc[1]=(bf16)vc.y; wc[2]=(bf16)vc.z; wc[3]=(bf16)vc.w;
  wd[0]=(bf16)vd.x; wd[1]=(bf16)vd.y; wd[2]=(bf16)vd.z; wd[3]=(bf16)vd.w;
  *(v4bf*)(oa + i * 4) = wa;  *(v4bf*)(ob + i * 4) = wb;
  *(v4bf*)(oc + i * 4) = wc;  *(v4bf*)(od + i * 4) = wd;
}

// ---------------------------------------------------------------------------
// LayerNorm over D=1024. One 256-thread block per row, 4 elems/thread.
// outb != nullptr -> write bf16 (pre-attention LN1); else write fp32 to outf.
// ---------------------------------------------------------------------------
__global__ __launch_bounds__(256) void ln_kernel(
    const float* __restrict__ in, const float* __restrict__ g,
    const float* __restrict__ bb, float* __restrict__ outf,
    bf16* __restrict__ outb) {
  __shared__ float rs[256], rq[256];
  int row = blockIdx.x, tid = threadIdx.x;
  float4 v = ((const float4*)(in + (size_t)row * D_))[tid];
  float s  = v.x + v.y + v.z + v.w;
  float sq = v.x*v.x + v.y*v.y + v.z*v.z + v.w*v.w;
  rs[tid] = s; rq[tid] = sq;
  __syncthreads();
#pragma unroll
  for (int off = 128; off > 0; off >>= 1) {
    if (tid < off) { rs[tid] += rs[tid + off]; rq[tid] += rq[tid + off]; }
    __syncthreads();
  }
  float mu   = rs[0] * (1.0f / D_);
  float var  = rq[0] * (1.0f / D_) - mu * mu;
  float rstd = rsqrtf(var + 1e-5f);
  float4 gg = ((const float4*)g)[tid];
  float4 b4 = ((const float4*)bb)[tid];
  float o0 = (v.x - mu) * rstd * gg.x + b4.x;
  float o1 = (v.y - mu) * rstd * gg.y + b4.y;
  float o2 = (v.z - mu) * rstd * gg.z + b4.z;
  float o3 = (v.w - mu) * rstd * gg.w + b4.w;
  size_t o = (size_t)row * D_ + tid * 4;
  if (outb) {
    v4bf w; w[0]=(bf16)o0; w[1]=(bf16)o1; w[2]=(bf16)o2; w[3]=(bf16)o3;
    *(v4bf*)(outb + o) = w;
  } else {
    outf[o+0]=o0; outf[o+1]=o1; outf[o+2]=o2; outf[o+3]=o3;
  }
}

// ---------------------------------------------------------------------------
// Kernel 3: fused Q/K/V projection GEMM.  y = xn @ W^T + b.
// Each wave: 16 rows x 64 cols.  Ping-pong double buffering with a 2x-unrolled
// steady state: fragments flow straight from loads into WMMA operands (no
// rotation copies, no forced loadcnt drain at the back-edge).
// Q gets the 1/sqrt(DH)=0.125 softmax scale folded in.  V stored transposed
// [bh, dh, s] so attention's P*V B-fragments are contiguous.
// ---------------------------------------------------------------------------
__global__ __launch_bounds__(256) void qkv_kernel(
    const bf16* __restrict__ xn,
    const bf16* __restrict__ wq, const bf16* __restrict__ wk, const bf16* __restrict__ wv,
    const float* __restrict__ bq, const float* __restrict__ bk, const float* __restrict__ bv,
    bf16* __restrict__ qO, bf16* __restrict__ kO, bf16* __restrict__ vTO) {
  int wave = blockIdx.x * (blockDim.x >> 5) + (threadIdx.x >> 5);  // 0..12287
  int lane = threadIdx.x & 31;
  int ng = wave % 48;        // 48 n-groups of 64 over 3*1024 outputs
  int mt = wave / 48;        // 256 row tiles of 16
  int row0 = mt * 16;
  int nall = ng * 64;
  int mat  = nall >> 10;     // 0=Q 1=K 2=V
  int ncol = nall & 1023;
  const bf16*  w    = (mat == 0) ? wq : (mat == 1) ? wk : wv;
  const float* bias = (mat == 0) ? bq : (mat == 1) ? bk : bv;
  const bf16* abase = xn + (size_t)row0 * D_;

  v8f acc[4] = {};
  Frags f0 = load_frags(abase, w, ncol, 0);
  Frags f1;
  for (int k0 = 0; k0 < D_ - 64; k0 += 64) {
    f1 = load_frags(abase, w, ncol, k0 + 32);   // in flight under f0's WMMAs
#pragma unroll
    for (int t = 0; t < 4; ++t) acc[t] = wmma_bf(f0.a, f0.b[t], acc[t]);
    f0 = load_frags(abase, w, ncol, k0 + 64);   // in flight under f1's WMMAs
#pragma unroll
    for (int t = 0; t < 4; ++t) acc[t] = wmma_bf(f1.a, f1.b[t], acc[t]);
  }
  f1 = load_frags(abase, w, ncol, D_ - 32);
#pragma unroll
  for (int t = 0; t < 4; ++t) acc[t] = wmma_bf(f0.a, f0.b[t], acc[t]);
#pragma unroll
  for (int t = 0; t < 4; ++t) acc[t] = wmma_bf(f1.a, f1.b[t], acc[t]);

  // ---- epilogue: bias, scale, scatter to head-major / transposed layouts ----
  int nlo = lane & 15, hi = lane >> 4;
#pragma unroll
  for (int t = 0; t < 4; ++t) {
#pragma unroll
    for (int e = 0; e < 8; ++e) {
      int nn = ncol + t * 16 + nlo;
      int r  = row0 + e + 8 * hi;            // flat row in [0, 4096)
      float val = acc[t][e] + bias[nn];
      int b = r >> 11, s = r & (S_ - 1);
      int h = nn >> 6, dh = nn & (DH_ - 1);
      int bh = b * H_ + h;
      if (mat == 0)      qO [((size_t)bh * S_ + s) * DH_ + dh] = (bf16)(val * 0.125f);
      else if (mat == 1) kO [((size_t)bh * S_ + s) * DH_ + dh] = (bf16)val;
      else               vTO[((size_t)bh * DH_ + dh) * S_ + s] = (bf16)val;
    }
  }
}

// ---------------------------------------------------------------------------
// Kernel 4: causal flash attention. One wave (= one workgroup, so divergent
// causal trip counts can't deadlock a barrier) owns 16 query rows of one
// (b,h).  Per 32-key block: all K/V fragment loads issued up front, two score
// WMMA pairs, then ONE combined online-softmax update (half the shuffle
// reductions), LDS C->A transpose of P fenced with s_wait_dscnt, 4 P*V WMMAs.
// ---------------------------------------------------------------------------
__global__ __launch_bounds__(32) void attn_kernel(
    const bf16* __restrict__ q, const bf16* __restrict__ k,
    const bf16* __restrict__ vT, bf16* __restrict__ ctx) {
  __shared__ __align__(16) bf16 pbuf[16 * 32];
  int wid  = blockIdx.x;                 // 0..4095
  int qt   = wid & (S_ / 16 - 1);        // query tile 0..127
  int bh   = wid >> 7;                   // 0..31
  int lane = threadIdx.x & 31;
  int nlo  = lane & 15, hi = lane >> 4;

  const bf16* qbase = q + ((size_t)bh * S_ + qt * 16) * DH_;
  v16bf A0 = load_A_frag(qbase,      DH_);   // K-dims 0..31
  v16bf A1 = load_A_frag(qbase + 32, DH_);   // K-dims 32..63

  float m[8], l[8];
  v8f acc[4] = {};
#pragma unroll
  for (int e = 0; e < 8; ++e) { m[e] = -INFINITY; l[e] = 0.0f; }

  const int qmax = qt * 16 + 15;
  for (int kb = 0; kb <= qmax; kb += 32) {
    bool has2 = (kb + 16 <= qmax);       // second 16-key sub-tile live?
    // ---- issue ALL loads for this 32-key block up front ----
    const bf16* kp0 = k + ((size_t)bh * S_ + kb) * DH_;
    v16bf Bk0a = load_B_frag(kp0,      DH_);
    v16bf Bk0b = load_B_frag(kp0 + 32, DH_);
    v16bf Bk1a, Bk1b;
    if (has2) {
      const bf16* kp1 = kp0 + 16 * DH_;
      Bk1a = load_B_frag(kp1,      DH_);
      Bk1b = load_B_frag(kp1 + 32, DH_);
    }
    v16bf Bv[4];
#pragma unroll
    for (int t = 0; t < 4; ++t)
      Bv[t] = load_B_frag(vT + ((size_t)bh * DH_ + t * 16) * S_ + kb, S_);

    // ---- scores: two 16x16 tiles over DH=64 (2 WMMAs each) ----
    v8f s0 = {};
    s0 = wmma_bf(A0, Bk0a, s0);
    s0 = wmma_bf(A1, Bk0b, s0);
    v8f s1 = {};
    if (has2) {
      s1 = wmma_bf(A0, Bk1a, s1);
      s1 = wmma_bf(A1, Bk1b, s1);
    }

    // ---- combined online softmax over the 32-key block ----
    int key0 = kb + nlo;
    int key1 = kb + 16 + nlo;
#pragma unroll
    for (int e = 0; e < 8; ++e) {
      int qrow = qt * 16 + e + 8 * hi;
      float sv0 = (key0 <= qrow)          ? s0[e] : -INFINITY;  // causal mask
      float sv1 = (has2 && key1 <= qrow)  ? s1[e] : -INFINITY;
      float rm = fmaxf(sv0, sv1);          // row-max over 16 lanes per half
#pragma unroll
      for (int off = 1; off < 16; off <<= 1)
        rm = fmaxf(rm, __shfl_xor(rm, off, 32));
      float mn    = fmaxf(m[e], rm);
      float alpha = __expf(m[e] - mn);     // -inf -> 0 on first block
      float pv0   = __expf(sv0 - mn);      // masked -> 0
      float pv1   = __expf(sv1 - mn);
      float rsum  = pv0 + pv1;
#pragma unroll
      for (int off = 1; off < 16; off <<= 1)
        rsum += __shfl_xor(rsum, off, 32);
      l[e] = l[e] * alpha + rsum;
      m[e] = mn;
#pragma unroll
      for (int t = 0; t < 4; ++t) acc[t][e] *= alpha;
      int prow = (e + 8 * hi) * 32;
      pbuf[prow + nlo]      = (bf16)pv0;   // C-layout -> row-major P in LDS
      pbuf[prow + 16 + nlo] = (bf16)pv1;
    }

    // wave-private LDS: order writes (all lanes) before A-layout reads
    __builtin_amdgcn_wave_barrier();
    asm volatile("s_wait_dscnt 0" ::: "memory");
    v16bf Ap = load_A_frag(pbuf, 32);      // P as 16x32 A-fragment
    __builtin_amdgcn_wave_barrier();
    asm volatile("s_wait_dscnt 0" ::: "memory");

    // ---- ctx += P(16x32) x V(32x64); B(k=key,n=dh) = vT[dh][key] ----
#pragma unroll
    for (int t = 0; t < 4; ++t)
      acc[t] = wmma_bf(Ap, Bv[t], acc[t]);
  }

  int b = bh >> 4, h = bh & 15;
#pragma unroll
  for (int t = 0; t < 4; ++t)
#pragma unroll
    for (int e = 0; e < 8; ++e) {
      int srow = qt * 16 + e + 8 * hi;
      float o = acc[t][e] / l[e];
      ctx[((size_t)(b * S_ + srow)) * D_ + h * DH_ + t * 16 + nlo] = (bf16)o;
    }
}

// ---------------------------------------------------------------------------
// Kernel 5: output projection + bias + residual: y = ctx @ Wo^T + bo + x (fp32)
// Same ping-pong pipeline as qkv_kernel.
// ---------------------------------------------------------------------------
__global__ __launch_bounds__(256) void proj_kernel(
    const bf16* __restrict__ ctx, const bf16* __restrict__ wo,
    const float* __restrict__ bo, const float* __restrict__ x,
    float* __restrict__ y) {
  int wave = blockIdx.x * (blockDim.x >> 5) + (threadIdx.x >> 5);  // 0..4095
  int lane = threadIdx.x & 31;
  int ng = wave & 15;
  int mt = wave >> 4;
  int row0 = mt * 16, ncol = ng * 64;
  const bf16* abase = ctx + (size_t)row0 * D_;

  v8f acc[4] = {};
  Frags f0 = load_frags(abase, wo, ncol, 0);
  Frags f1;
  for (int k0 = 0; k0 < D_ - 64; k0 += 64) {
    f1 = load_frags(abase, wo, ncol, k0 + 32);
#pragma unroll
    for (int t = 0; t < 4; ++t) acc[t] = wmma_bf(f0.a, f0.b[t], acc[t]);
    f0 = load_frags(abase, wo, ncol, k0 + 64);
#pragma unroll
    for (int t = 0; t < 4; ++t) acc[t] = wmma_bf(f1.a, f1.b[t], acc[t]);
  }
  f1 = load_frags(abase, wo, ncol, D_ - 32);
#pragma unroll
  for (int t = 0; t < 4; ++t) acc[t] = wmma_bf(f0.a, f0.b[t], acc[t]);
#pragma unroll
  for (int t = 0; t < 4; ++t) acc[t] = wmma_bf(f1.a, f1.b[t], acc[t]);

  int nlo = lane & 15, hi = lane >> 4;
#pragma unroll
  for (int t = 0; t < 4; ++t)
#pragma unroll
    for (int e = 0; e < 8; ++e) {
      int nn = ncol + t * 16 + nlo;
      int r  = row0 + e + 8 * hi;
      size_t idx = (size_t)r * D_ + nn;
      y[idx] = acc[t][e] + bo[nn] + x[idx];
    }
}

// ---------------------------------------------------------------------------
// Host-side launcher
// ---------------------------------------------------------------------------
extern "C" void kernel_launch(void* const* d_in, const int* in_sizes, int n_in,
                              void* d_out, int out_size, void* d_ws, size_t ws_size,
                              hipStream_t stream) {
  const float* x  = (const float*)d_in[0];
  const float* Wq = (const float*)d_in[1];  const float* bq = (const float*)d_in[2];
  const float* Wk = (const float*)d_in[3];  const float* bk = (const float*)d_in[4];
  const float* Wv = (const float*)d_in[5];  const float* bv = (const float*)d_in[6];
  const float* Wo = (const float*)d_in[7];  const float* bo = (const float*)d_in[8];
  const float* g1 = (const float*)d_in[9];  const float* b1 = (const float*)d_in[10];
  const float* g2 = (const float*)d_in[11]; const float* b2 = (const float*)d_in[12];

  // Workspace carve (all 256B aligned). Total ~64 MB.
  char* wsp = (char*)d_ws;
  auto carve = [&](size_t bytes) -> char* {
    char* p = wsp;
    wsp += (bytes + 255) & ~(size_t)255;
    return p;
  };
  bf16* xn   = (bf16*)carve((size_t)ROWS * D_ * 2);
  bf16* wqb  = (bf16*)carve((size_t)D_ * D_ * 2);
  bf16* wkb  = (bf16*)carve((size_t)D_ * D_ * 2);
  bf16* wvb  = (bf16*)carve((size_t)D_ * D_ * 2);
  bf16* wob  = (bf16*)carve((size_t)D_ * D_ * 2);
  bf16* qb   = (bf16*)carve((size_t)ROWS * D_ * 2);
  bf16* kb   = (bf16*)carve((size_t)ROWS * D_ * 2);
  bf16* vTb  = (bf16*)carve((size_t)ROWS * D_ * 2);
  bf16* ctxb = (bf16*)carve((size_t)ROWS * D_ * 2);
  float* y   = (float*)carve((size_t)ROWS * D_ * 4);

  // 1) weights -> bf16
  cvtw_kernel<<<(D_ * D_ / 4) / 256, 256, 0, stream>>>(Wq, Wk, Wv, Wo,
                                                       wqb, wkb, wvb, wob);
  // 2) LN1(x) -> bf16 xn
  ln_kernel<<<ROWS, 256, 0, stream>>>(x, g1, b1, nullptr, xn);
  // 3) Q/K/V projections (12288 waves, 8 waves/block)
  qkv_kernel<<<1536, 256, 0, stream>>>(xn, wqb, wkb, wvb, bq, bk, bv, qb, kb, vTb);
  // 4) causal flash attention (4096 single-wave workgroups)
  attn_kernel<<<B_ * H_ * (S_ / 16), 32, 0, stream>>>(qb, kb, vTb, ctxb);
  // 5) output projection + residual (4096 waves, 8 waves/block)
  proj_kernel<<<512, 256, 0, stream>>>(ctxb, wob, bo, x, y);
  // 6) LN2 -> d_out (fp32)
  ln_kernel<<<ROWS, 256, 0, stream>>>(y, g2, b2, (float*)d_out, nullptr);
}
